// DenoisingUNet_7035156431277
// MI455X (gfx1250) — matrix operand order repaired and checked by reading, not compiled, not run on hardware
//
#include <hip/hip_runtime.h>
#include <hip/hip_bf16.h>
#include <cstddef>

// ---------------------------------------------------------------------------
// DenoisingUNet (GAT U-Net) forward for MI455X / gfx1250.
//  * GEMMs: v_wmma_f32_16x16x32_bf16, A tiles staged to LDS by the Tensor
//    Data Mover (tensor_load_to_lds + s_wait_tensorcnt), double-buffered.
//  * Weights pre-converted once per launch to bf16, transposed [O,K] so a
//    B fragment is two contiguous 16B loads per lane (no hot-loop cvt).
//  * Attention / segment ops in fp32 with atomics (memory-bound part).
// Input-leaf ordering assumption (JAX pytree flatten, dict keys sorted):
//   0:x_t 1:time_embed 2:src 3:dst
//   4..15  down[0..2]   each {W, attn_l, attn_r, bias}
//   16..27 mlp_in       {a1,a_out,in.b,in.w,l1.b,l1.w,l2.b,l2.w,ln_b,ln_g,out.b,out.w}
//   28..39 mlp_middle   (same layout)
//   40..51 mlp_out      (same layout)
//   52..63 up[0..2]     each {W, attn_l, attn_r, bias}  (up[0],up[1]=4 heads; up[2]=1)
// ---------------------------------------------------------------------------

#define NODES 50000
#define EDGES 800000
#define HID   256

typedef __bf16        v16bf __attribute__((ext_vector_type(16)));
typedef float         v8f   __attribute__((ext_vector_type(8)));
typedef unsigned int  u32x4 __attribute__((ext_vector_type(4)));
typedef int           i32x8 __attribute__((ext_vector_type(8)));
typedef int           i32x4 __attribute__((ext_vector_type(4)));

// ---------------------------------------------------------------------------
// Weight convert: Wt(bf16)[O][K] = W(f32)[K][O], packed as dwords.
// ---------------------------------------------------------------------------
__global__ void k_wt_convert(const float* __restrict__ W, unsigned* __restrict__ Wt2,
                             int K, int O)
{
    const int idx = blockIdx.x * blockDim.x + threadIdx.x;   // o*(K/2) + kp
    const int tot = O * (K >> 1);
    if (idx >= tot) return;
    const int o = idx / (K >> 1), kp = idx % (K >> 1);
    const int k = kp * 2;
    union { __bf16 h; unsigned short s; } a, b;
    a.h = (__bf16)W[(size_t)k * O + o];
    b.h = (__bf16)W[(size_t)(k + 1) * O + o];
    Wt2[idx] = ((unsigned)b.s << 16) | (unsigned)a.s;
}

// ---------------------------------------------------------------------------
// WMMA GEMM: C[N,O] = act( A[N,K] @ W[K,O] + bias [+ res] )
// Block = 256 thr = 8 waves; block tile = 16 rows x 128 cols (1 col-tile/wave).
// A tile (16x32 f32) staged to LDS by the TDM, double-buffered; issued by
// wave 0, published to the workgroup with s_wait_tensorcnt + barrier.
// Requires N%16==0, K%32==0, O%128==0 (true for all shapes here).
// ---------------------------------------------------------------------------
__global__ __launch_bounds__(256)
void k_gemm_wmma(const float* __restrict__ A, const unsigned short* __restrict__ Wt,
                 const float* __restrict__ bias, const float* __restrict__ res,
                 const float* __restrict__ aprelu, float* __restrict__ C,
                 int N, int K, int O)
{
    __shared__ float smem[2][16 * 32];          // 2 x 2KB A tiles (f32)

    const int wave = threadIdx.x >> 5;
    const int lane = threadIdx.x & 31;
    const int rt   = blockIdx.x;                // 16-row tile
    const int ct   = blockIdx.y * 8 + wave;     // 16-col tile (always in range)
    const int row  = lane & 15;
    const int col  = ct * 16 + (lane & 15);
    const int hi   = lane >> 4;                 // lane half (0/1)

    const unsigned long long abase = (unsigned long long)(const void*)A;

    // Issue a TDM 2-D tile load: A[rt*16 .. +15, k0 .. k0+31] -> LDS buffer.
    // D# built per CDNA5 ISA 8.3/8.4 (group0: count/lds/global/type,
    // group1: data_size=4B, dims, tile 32x16, stride K). Static LDS assumed
    // to start at group-segment offset 0.
    auto tdm_issue = [&](int buf, int k0) {
        const unsigned long long ga =
            abase + ((((unsigned long long)rt * 16ull) * (unsigned)K + (unsigned)k0) << 2);
        u32x4 g0;
        g0.x = 1u;                                           // count=1
        g0.y = (unsigned)(buf * 2048);                       // lds_addr (bytes)
        g0.z = (unsigned)ga;                                 // global_addr[31:0]
        g0.w = (unsigned)((ga >> 32) & 0x01FFFFFFu) | (2u << 30);  // addr[56:32]|type=2
        i32x8 g1;
        g1[0] = 0x00020000;                                  // data_size=2 (4 bytes)
        g1[1] = (int)(((unsigned)K & 0xFFFFu) << 16);        // tensor_dim0[15:0]
        g1[2] = (int)((((unsigned)K >> 16) & 0xFFFFu) |
                      (((unsigned)N & 0xFFFFu) << 16));      // dim0 hi | dim1 lo
        g1[3] = (int)((((unsigned)N >> 16) & 0xFFFFu) |
                      (32u << 16));                          // dim1 hi | tile_dim0=32
        g1[4] = 16;                                          // tile_dim1=16
        g1[5] = (int)(unsigned)K;                            // tensor_dim0_stride lo
        g1[6] = 0;                                           // stride hi | dim1_stride lo
        g1[7] = 0;
        const i32x4 z4 = { 0, 0, 0, 0 };
        const i32x8 z8 = { 0, 0, 0, 0, 0, 0, 0, 0 };
        __builtin_amdgcn_tensor_load_to_lds(g0, g1, z4, z4, z8, 0);
    };

    if (wave == 0) tdm_issue(0, 0);

    v8f acc = {};
    int buf = 0;
    for (int k0 = 0; k0 < K; k0 += 32) {
        if (wave == 0) __builtin_amdgcn_s_wait_tensorcnt(0);
        __syncthreads();
        // LDS is written behind the compiler's back by the TDM:
        asm volatile("" : : "r"(&smem[0][0]) : "memory");
        if (wave == 0 && (k0 + 32) < K) tdm_issue(buf ^ 1, k0 + 32);

        // ---- A fragment (this kernel's k order, same as B): lane row = L&15,
        // elements {hi*16..hi*16+15} (contiguous k); convert f32->bf16.
        const float* arow = &smem[buf][row * 32];
        const float4 r0 = *(const float4*)(arow + hi * 16);
        const float4 r1 = *(const float4*)(arow + hi * 16 + 4);
        const float4 r2 = *(const float4*)(arow + hi * 16 + 8);
        const float4 r3 = *(const float4*)(arow + hi * 16 + 12);
        v16bf af;
        af[0]  = (__bf16)r0.x; af[1]  = (__bf16)r0.y;
        af[2]  = (__bf16)r0.z; af[3]  = (__bf16)r0.w;
        af[4]  = (__bf16)r1.x; af[5]  = (__bf16)r1.y;
        af[6]  = (__bf16)r1.z; af[7]  = (__bf16)r1.w;
        af[8]  = (__bf16)r2.x; af[9]  = (__bf16)r2.y;
        af[10] = (__bf16)r2.z; af[11] = (__bf16)r2.w;
        af[12] = (__bf16)r3.x; af[13] = (__bf16)r3.y;
        af[14] = (__bf16)r3.z; af[15] = (__bf16)r3.w;

        // ---- B fragment: Wt[col][k0+hi*16 .. +15] = 16 contiguous bf16.
        const uint4* wp = (const uint4*)(Wt + (size_t)col * K + k0 + hi * 16);
        union { v16bf v; uint4 u[2]; } bfu;
        bfu.u[0] = wp[0];
        bfu.u[1] = wp[1];
        if ((k0 + 32) < K)
            __builtin_prefetch(Wt + (size_t)col * K + (k0 + 32) + hi * 16, 0, 3);

        acc = __builtin_amdgcn_wmma_f32_16x16x32_bf16(
                  false, af, false, bfu.v, (short)0, acc, false, false);
        buf ^= 1;
    }

    const float bv = bias ? bias[col] : 0.0f;
    const float ap = aprelu ? *aprelu : 0.0f;
    #pragma unroll
    for (int v = 0; v < 8; ++v) {
        const int m = rt * 16 + hi * 8 + v;   // C/D layout: VGPR v -> M=hi*8+v
        float c = acc[v] + bv;
        if (res)    c += res[(size_t)m * O + col];
        if (aprelu) c  = (c >= 0.0f) ? c : ap * c;
        C[(size_t)m * O + col] = c;
    }
}

// ---------------------------------------------------------------------------
// Row-wise LayerNorm + PReLU, one wave32 per row.
// ---------------------------------------------------------------------------
__global__ __launch_bounds__(256)
void k_ln_prelu(float* __restrict__ X, const float* __restrict__ g,
                const float* __restrict__ b, const float* __restrict__ ap,
                int N, int H)
{
    const int wave = threadIdx.x >> 5, lane = threadIdx.x & 31;
    const int row = blockIdx.x * 8 + wave;
    if (row >= N) return;
    float* x = X + (size_t)row * H;
    float s = 0.0f, s2 = 0.0f;
    for (int i = lane; i < H; i += 32) { const float v = x[i]; s += v; s2 += v * v; }
    #pragma unroll
    for (int off = 16; off >= 1; off >>= 1) {
        s  += __shfl_xor(s,  off, 32);
        s2 += __shfl_xor(s2, off, 32);
    }
    const float m   = s / (float)H;
    const float var = s2 / (float)H - m * m;
    const float inv = rsqrtf(var + 1e-5f);
    const float a   = *ap;
    for (int i = lane; i < H; i += 32) {
        float v = (x[i] - m) * inv * g[i] + b[i];
        x[i] = (v >= 0.0f) ? v : a * v;
    }
}

__global__ void k_add3(float* __restrict__ out, const float* __restrict__ a,
                       const float* __restrict__ b, const float* __restrict__ c,
                       long long n)
{
    const long long i = (long long)blockIdx.x * blockDim.x + threadIdx.x;
    if (i >= n) return;
    float v = a[i] + b[i];
    if (c) v += c[i];
    out[i] = v;
}

__global__ void k_fill(float* __restrict__ p, float v, long long n)
{
    const long long i = (long long)blockIdx.x * blockDim.x + threadIdx.x;
    if (i < n) p[i] = v;
}

// ---------------------------------------------------------------------------
// GAT attention coefficients: el/er[n,h] = sum_d feat[n,h,d]*attn_{l,r}[h,d]
// ---------------------------------------------------------------------------
__global__ void k_attn(const float* __restrict__ feat, const float* __restrict__ al,
                       const float* __restrict__ ar, float* __restrict__ el,
                       float* __restrict__ er, int N, int nh, int dh)
{
    const int idx = blockIdx.x * blockDim.x + threadIdx.x;   // node*nh + h
    if (idx >= N * nh) return;
    const int n = idx / nh, h = idx % nh;
    const float* f = feat + (size_t)n * nh * dh + (size_t)h * dh;
    const float* L = al + (size_t)h * dh;
    const float* R = ar + (size_t)h * dh;
    float sl = 0.0f, sr = 0.0f;
    for (int d = 0; d < dh; ++d) { const float v = f[d]; sl += v * L[d]; sr += v * R[d]; }
    el[idx] = sl; er[idx] = sr;
}

// float atomic max via sign-aware int/uint atomics
__device__ inline void atomicMaxF(float* addr, float val)
{
    if (val >= 0.0f) atomicMax((int*)addr, __float_as_int(val));
    else             atomicMin((unsigned int*)addr, __float_as_uint(val));
}

__global__ void k_edge_e(const int* __restrict__ src, const int* __restrict__ dst,
                         const float* __restrict__ el, const float* __restrict__ er,
                         float* __restrict__ e, float* __restrict__ emax,
                         int E, int nh)
{
    const int idx = blockIdx.x * blockDim.x + threadIdx.x;   // edge*nh + h
    if (idx >= E * nh) return;
    const int ed = idx / nh, h = idx % nh;
    const int s = src[ed], d = dst[ed];
    float v = el[s * nh + h] + er[d * nh + h];
    v = (v >= 0.0f) ? v : 0.2f * v;
    e[idx] = v;
    atomicMaxF(&emax[d * nh + h], v);
}

__global__ void k_edge_alpha(const int* __restrict__ dst, float* __restrict__ e,
                             const float* __restrict__ emax, float* __restrict__ denom,
                             int E, int nh)
{
    const int idx = blockIdx.x * blockDim.x + threadIdx.x;
    if (idx >= E * nh) return;
    const int ed = idx / nh, h = idx % nh;
    const int d = dst[ed];
    const float a = expf(e[idx] - emax[d * nh + h]);
    e[idx] = a;
    atomicAdd(&denom[d * nh + h], a);
}

// out[dst, c] += feat[src, c] * alpha/denom ; one block (256 thr) per edge
__global__ __launch_bounds__(256)
void k_edge_aggr(const int* __restrict__ src, const int* __restrict__ dst,
                 const float* __restrict__ feat, const float* __restrict__ alpha,
                 const float* __restrict__ denom, float* __restrict__ out,
                 int nh, int dh)
{
    const int ed = blockIdx.x;
    const int c  = threadIdx.x;          // 0..255 = nh*dh
    const int h  = c / dh;
    const int s = src[ed], d = dst[ed];
    const float a = alpha[(size_t)ed * nh + h] / fmaxf(denom[d * nh + h], 1e-9f);
    atomicAdd(&out[(size_t)d * HID + c], feat[(size_t)s * HID + c] * a);
}

__global__ __launch_bounds__(256)
void k_finalize(float* __restrict__ gout, const float* __restrict__ bias,
                float* __restrict__ dout, int ostride, int coff)
{
    const int n = blockIdx.x;
    const int c = threadIdx.x;           // 0..255
    const float v = gout[(size_t)n * HID + c] + bias[c];
    gout[(size_t)n * HID + c] = v;
    if (dout) dout[(size_t)n * ostride + coff + c] = v;
}

// ---------------------------------------------------------------------------
// Host-side helpers
// ---------------------------------------------------------------------------
struct MlpP {
    const float *a1, *a_out, *in_b, *in_w, *l1_b, *l1_w, *l2_b, *l2_w,
                *ln_b, *ln_g, *out_b, *out_w;
};
struct MlpW { const unsigned short *in_w, *l1_w, *l2_w, *out_w; };
struct GatP { const float *W, *al, *ar, *bias; };

static inline MlpP mlp_from(void* const* d_in, int base) {
    MlpP p;
    p.a1   = (const float*)d_in[base + 0];  p.a_out = (const float*)d_in[base + 1];
    p.in_b = (const float*)d_in[base + 2];  p.in_w  = (const float*)d_in[base + 3];
    p.l1_b = (const float*)d_in[base + 4];  p.l1_w  = (const float*)d_in[base + 5];
    p.l2_b = (const float*)d_in[base + 6];  p.l2_w  = (const float*)d_in[base + 7];
    p.ln_b = (const float*)d_in[base + 8];  p.ln_g  = (const float*)d_in[base + 9];
    p.out_b= (const float*)d_in[base +10];  p.out_w = (const float*)d_in[base +11];
    return p;
}
static inline GatP gat_from(void* const* d_in, int base) {
    GatP p;
    p.W  = (const float*)d_in[base + 0]; p.al   = (const float*)d_in[base + 1];
    p.ar = (const float*)d_in[base + 2]; p.bias = (const float*)d_in[base + 3];
    return p;
}

static void gemm(const float* A, const unsigned short* Wt, const float* bias,
                 const float* res, const float* ap, float* C, int K, int O,
                 hipStream_t s)
{
    dim3 g(NODES / 16, O / 128);
    k_gemm_wmma<<<g, 256, 0, s>>>(A, Wt, bias, res, ap, C, NODES, K, O);
}

static void mlp_block(const float* xin, int K, int H, int O, const MlpP& p,
                      const MlpW& w, float* xbuf, float* hbuf, float* outp,
                      hipStream_t s)
{
    gemm(xin,  w.in_w, p.in_b, nullptr, nullptr, xbuf, K, H, s);   // x = xin@in+b
    gemm(xbuf, w.l1_w, p.l1_b, nullptr, nullptr, hbuf, H, H, s);   // h = x@l1+b
    k_ln_prelu<<<(NODES + 7) / 8, 256, 0, s>>>(hbuf, p.ln_g, p.ln_b, p.a1, NODES, H);
    gemm(hbuf, w.l2_w, p.l2_b, xbuf, nullptr, xbuf, H, H, s);      // x = x + h@l2+b
    gemm(xbuf, w.out_w, p.out_b, nullptr, p.a_out, outp, H, O, s); // prelu(x@out+b)
}

static void gat_layer(const float* xin, const GatP& p, const unsigned short* wtW,
                      int nh, float* feat, float* gout,
                      float* el, float* er, float* emax, float* denom, float* ebuf,
                      const int* src, const int* dst,
                      float* dout_region, int coff, hipStream_t s)
{
    const int dh = HID / nh;
    gemm(xin, wtW, nullptr, nullptr, nullptr, feat, HID, HID, s);
    k_attn<<<(NODES * nh + 255) / 256, 256, 0, s>>>(feat, p.al, p.ar, el, er, NODES, nh, dh);
    k_fill<<<(NODES * nh + 255) / 256, 256, 0, s>>>(emax, -3.0e38f, (long long)NODES * nh);
    k_fill<<<(NODES * nh + 255) / 256, 256, 0, s>>>(denom, 0.0f, (long long)NODES * nh);
    k_fill<<<((long long)NODES * HID + 255) / 256, 256, 0, s>>>(gout, 0.0f, (long long)NODES * HID);
    k_edge_e<<<(EDGES * nh + 255) / 256, 256, 0, s>>>(src, dst, el, er, ebuf, emax, EDGES, nh);
    k_edge_alpha<<<(EDGES * nh + 255) / 256, 256, 0, s>>>(dst, ebuf, emax, denom, EDGES, nh);
    k_edge_aggr<<<EDGES, 256, 0, s>>>(src, dst, feat, ebuf, denom, gout, nh, dh);
    k_finalize<<<NODES, 256, 0, s>>>(gout, p.bias, dout_region, 3 * HID, coff);
}

// ---------------------------------------------------------------------------
extern "C" void kernel_launch(void* const* d_in, const int* in_sizes, int n_in,
                              void* d_out, int out_size, void* d_ws, size_t ws_size,
                              hipStream_t stream)
{
    (void)in_sizes; (void)n_in; (void)out_size; (void)ws_size;

    const float* X_T  = (const float*)d_in[0];
    const float* TEMB = (const float*)d_in[1];
    const int*   SRC  = (const int*)d_in[2];
    const int*   DST  = (const int*)d_in[3];

    GatP down[3] = { gat_from(d_in, 4), gat_from(d_in, 8), gat_from(d_in, 12) };
    MlpP mlp_in  = mlp_from(d_in, 16);
    MlpP mlp_mid = mlp_from(d_in, 28);
    MlpP mlp_out = mlp_from(d_in, 40);
    GatP up[3]   = { gat_from(d_in, 52), gat_from(d_in, 56), gat_from(d_in, 60) };
    const int up_nh[3] = { 4, 4, 1 };

    float* OUT0 = (float*)d_out;                        // [N,128]
    float* OUT1 = OUT0 + (size_t)NODES * 128;           // [N,768] concat(out_hidden)

    // ---- workspace carve (floats) ----
    float* ws = (float*)d_ws;
    size_t o = 0;
    float* bufX = ws + o; o += (size_t)NODES * 512;     // mlp x / up-GAT accumulator
    float* bufH = ws + o; o += (size_t)NODES * 512;     // mlp h / GAT feat
    float* bufA = ws + o; o += (size_t)NODES * HID;     // current hidden
    float* bufB = ws + o; o += (size_t)NODES * HID;     // GAT input (h + skips)
    float* dh0  = ws + o; o += (size_t)NODES * HID;
    float* dh1  = ws + o; o += (size_t)NODES * HID;
    float* dh2  = ws + o; o += (size_t)NODES * HID;
    float* el   = ws + o; o += (size_t)NODES * 4;
    float* er   = ws + o; o += (size_t)NODES * 4;
    float* emax = ws + o; o += (size_t)NODES * 4;
    float* den  = ws + o; o += (size_t)NODES * 4;
    float* ebuf = ws + o; o += (size_t)EDGES * 4;       // e, then alpha
    unsigned short* wtp = (unsigned short*)(ws + o);    // bf16 weight pool (~2.9MB)

    // ---- pre-convert all GEMM weights to bf16, transposed [O,K] ----
    size_t wto = 0;
    auto conv = [&](const float* W, int K, int O) -> const unsigned short* {
        unsigned short* dst = wtp + wto;
        wto += (size_t)K * O;
        const int tot = O * (K / 2);
        k_wt_convert<<<(tot + 255) / 256, 256, 0, stream>>>(W, (unsigned*)dst, K, O);
        return dst;
    };
    MlpW w_in  = { conv(mlp_in.in_w, 128, 512),  conv(mlp_in.l1_w, 512, 512),
                   conv(mlp_in.l2_w, 512, 512),  conv(mlp_in.out_w, 512, HID) };
    MlpW w_mid = { conv(mlp_mid.in_w, HID, HID), conv(mlp_mid.l1_w, HID, HID),
                   conv(mlp_mid.l2_w, HID, HID), conv(mlp_mid.out_w, HID, HID) };
    MlpW w_out = { conv(mlp_out.in_w, HID, 128), conv(mlp_out.l1_w, 128, 128),
                   conv(mlp_out.l2_w, 128, 128), conv(mlp_out.out_w, 128, 128) };
    const unsigned short* w_down[3];
    const unsigned short* w_up[3];
    for (int l = 0; l < 3; ++l) w_down[l] = conv(down[l].W, HID, HID);
    for (int l = 0; l < 3; ++l) w_up[l]   = conv(up[l].W,   HID, HID);

    float* dhs[3] = { dh0, dh1, dh2 };

    // ---- encoder MLP: [N,128] -> [N,256]
    mlp_block(X_T, 128, 512, HID, mlp_in, w_in, bufX, bufH, bufA, stream);

    // ---- down GAT layers
    const float* hcur = bufA;
    for (int l = 0; l < 3; ++l) {
        k_add3<<<((long long)NODES * HID + 255) / 256, 256, 0, stream>>>(
            bufB, hcur, TEMB, nullptr, (long long)NODES * HID);
        gat_layer(bufB, down[l], w_down[l], 4, bufH, dhs[l], el, er, emax, den, ebuf,
                  SRC, DST, nullptr, 0, stream);
        hcur = dhs[l];
    }

    // ---- middle MLP
    mlp_block(dh2, HID, HID, HID, mlp_mid, w_mid, bufX, bufH, bufA, stream);
    hcur = bufA;

    // ---- up GAT layers (skip connections + out_hidden concat)
    for (int i = 0; i < 3; ++i) {
        k_add3<<<((long long)NODES * HID + 255) / 256, 256, 0, stream>>>(
            bufB, hcur, dhs[2 - i], TEMB, (long long)NODES * HID);
        gat_layer(bufB, up[i], w_up[i], up_nh[i], bufH, bufX, el, er, emax, den, ebuf,
                  SRC, DST, OUT1, i * HID, stream);
        hcur = bufX;
    }

    // ---- decoder MLP: [N,256] -> [N,128] into d_out
    mlp_block(bufX, HID, 128, 128, mlp_out, w_out, bufB, bufH, OUT0, stream);
}
